// SwinTransformerLayer_74431783240066
// MI455X (gfx1250) — matrix operand order repaired and checked by reading, compiled, not run on hardware
//
#include <hip/hip_runtime.h>
#include <hip/hip_bf16.h>

typedef _Float16 v16h __attribute__((ext_vector_type(16)));
typedef _Float16 v8h  __attribute__((ext_vector_type(8)));
typedef float    v8f  __attribute__((ext_vector_type(8)));

// ---------------- problem constants ----------------
constexpr int Zd = 8, Hd = 96, Wd = 192;
constexpr int WZ = 2, WH = 6, WW = 12;
constexpr int SZ = 1, SH = 3, SW = 6;
constexpr int NZ = 4, NH = 16, NW = 16;
constexpr int LW = 144;              // tokens per window
constexpr int Cc = 192;
constexpr int HEADS = 6, HD = 32;
constexpr int NWIN = NZ * NH * NW;   // 1024
constexpr long NT = (long)Zd * Hd * Wd;   // 147456 tokens
constexpr float SCALE = 0.17677669529663687f;   // 32^-0.5

__device__ __forceinline__ v16h cat16(v8h a, v8h b) {
  return __builtin_shufflevector(a, b, 0,1,2,3,4,5,6,7,8,9,10,11,12,13,14,15);
}

// window (win,t) -> flat token index (handles roll; same map for LN-in and proj-out)
__device__ __forceinline__ long win_token(int win, int t, int roll) {
  int nz = win >> 8, nh = (win >> 4) & 15, nw = win & 15;
  int wz = t / (WH * WW); int rem = t % (WH * WW);
  int wh = rem / WW, ww = rem % WW;
  int z = nz * WZ + wz, h = nh * WH + wh, w = nw * WW + ww;
  if (roll) {
    z = (z + SZ) & (Zd - 1);
    h += SH; if (h >= Hd) h -= Hd;
    w += SW; if (w >= Wd) w -= Wd;
  }
  return ((long)z * Hd + h) * Wd + w;
}

// ---------------- LayerNorm kernels (1 wave / token) ----------------
__global__ __launch_bounds__(256) void ln1_win_kernel(
    const float* __restrict__ x, const float* __restrict__ g,
    const float* __restrict__ b, _Float16* __restrict__ h16, int roll)
{
  const int d    = blockIdx.x * 8 + (threadIdx.x >> 5);   // dest (win,t) flat
  const int lane = threadIdx.x & 31;
  const int win  = d / LW, t = d % LW;
  const long tok = win_token(win, t, roll);
  const float* xs = x + tok * Cc;
  float vals[6]; float s = 0.f;
  #pragma unroll
  for (int j = 0; j < 6; ++j) { vals[j] = xs[lane + 32 * j]; s += vals[j]; }
  #pragma unroll
  for (int off = 16; off >= 1; off >>= 1) s += __shfl_xor(s, off, 32);
  const float mean = s * (1.f / 192.f);
  float vs = 0.f;
  #pragma unroll
  for (int j = 0; j < 6; ++j) { float dd = vals[j] - mean; vs += dd * dd; }
  #pragma unroll
  for (int off = 16; off >= 1; off >>= 1) vs += __shfl_xor(vs, off, 32);
  const float inv = rsqrtf(vs * (1.f / 192.f) + 1e-5f);
  _Float16* out = h16 + (long)d * Cc;
  #pragma unroll
  for (int j = 0; j < 6; ++j) {
    const int c = lane + 32 * j;
    out[c] = (_Float16)((vals[j] - mean) * inv * g[c] + b[c]);
  }
}

__global__ __launch_bounds__(256) void ln2_kernel(
    const float* __restrict__ x, const float* __restrict__ g,
    const float* __restrict__ b, _Float16* __restrict__ h16)
{
  const long d   = (long)blockIdx.x * 8 + (threadIdx.x >> 5);
  const int lane = threadIdx.x & 31;
  const float* xs = x + d * Cc;
  float vals[6]; float s = 0.f;
  #pragma unroll
  for (int j = 0; j < 6; ++j) { vals[j] = xs[lane + 32 * j]; s += vals[j]; }
  #pragma unroll
  for (int off = 16; off >= 1; off >>= 1) s += __shfl_xor(s, off, 32);
  const float mean = s * (1.f / 192.f);
  float vs = 0.f;
  #pragma unroll
  for (int j = 0; j < 6; ++j) { float dd = vals[j] - mean; vs += dd * dd; }
  #pragma unroll
  for (int off = 16; off >= 1; off >>= 1) vs += __shfl_xor(vs, off, 32);
  const float inv = rsqrtf(vs * (1.f / 192.f) + 1e-5f);
  _Float16* out = h16 + d * Cc;
  #pragma unroll
  for (int j = 0; j < 6; ++j) {
    const int c = lane + 32 * j;
    out[c] = (_Float16)((vals[j] - mean) * inv * g[c] + b[c]);
  }
}

// ---------------- shared WMMA GEMM mainloop: 128x64 tile, 8 waves -------------
// A: row-major MxK f16 (global); Bw: row-major KxNLD f32 (global, cvt->f16)
template<int K, int NLD>
__device__ __forceinline__ void gemm_mainloop(
    const _Float16* __restrict__ A, const float* __restrict__ Bw,
    int m0, int n0, _Float16* As, _Float16* Bs, v8f (&acc)[4])
{
  const int tid  = threadIdx.x;
  const int lane = tid & 31;
  const int wave = tid >> 5;
  const int half = lane >> 4;
  const int lm   = lane & 15;
  const int arow = wave * 16 + lm;
  const int lrow = tid >> 1;            // 0..127
  const int lseg = (tid & 1) * 16;      // 0 / 16 halves
  const int bkr  = tid & 31;            // B tile k row
  const int bnc  = (tid >> 5) * 8;      // B tile n base

  for (int k0 = 0; k0 < K; k0 += 32) {
    __syncthreads();
    { // A tile: 128 x 32 halves
      const _Float16* src = A + (long)(m0 + lrow) * K + k0 + lseg;
      uint4* dst = (uint4*)&As[lrow * 32 + lseg];
      dst[0] = ((const uint4*)src)[0];
      dst[1] = ((const uint4*)src)[1];
      if (k0 + 32 < K)
        __builtin_prefetch(src + 32, 0, 0);   // global_prefetch_b8
    }
    { // B tile: 32(k) x 64(n), stored transposed [n][k] in LDS
      const float* bsrc = Bw + (long)(k0 + bkr) * NLD + (n0 + bnc);
      #pragma unroll
      for (int j = 0; j < 8; ++j)
        Bs[(bnc + j) * 32 + bkr] = (_Float16)bsrc[j];
    }
    __syncthreads();
    v8h a0 = *(const v8h*)&As[arow * 32 + half * 8];
    v8h a1 = *(const v8h*)&As[arow * 32 + 16 + half * 8];
    v16h a = cat16(a0, a1);
    #pragma unroll
    for (int nt = 0; nt < 4; ++nt) {
      const int bn = nt * 16 + lm;
      v8h b0 = *(const v8h*)&Bs[bn * 32 + half * 8];
      v8h b1 = *(const v8h*)&Bs[bn * 32 + 16 + half * 8];
      v16h bb = cat16(b0, b1);
      acc[nt] = __builtin_amdgcn_wmma_f32_16x16x32_f16(
          false, a, false, bb, (short)0, acc[nt], false, false);
    }
  }
}

// ---------------- QKV GEMM: (NT,192)@(192,576)+b, scatter to q/k/v ------------
__global__ __launch_bounds__(256) void gemm_qkv_kernel(
    const _Float16* __restrict__ A, const float* __restrict__ W,
    const float* __restrict__ bias,
    _Float16* __restrict__ q16, _Float16* __restrict__ k16, _Float16* __restrict__ v16)
{
  __shared__ _Float16 As[128 * 32];
  __shared__ _Float16 Bs[64 * 32];
  const int m0 = blockIdx.x * 128, n0 = blockIdx.y * 64;
  v8f z8 = {0.f,0.f,0.f,0.f,0.f,0.f,0.f,0.f};
  v8f acc[4] = {z8, z8, z8, z8};
  gemm_mainloop<192, 576>(A, W, m0, n0, As, Bs, acc);
  const int tid = threadIdx.x, lane = tid & 31, wave = tid >> 5;
  const int half = lane >> 4, lm = lane & 15;
  #pragma unroll
  for (int nt = 0; nt < 4; ++nt)
    #pragma unroll
    for (int r = 0; r < 8; ++r) {
      const long gm = m0 + wave * 16 + r + 8 * half;
      const int  gn = n0 + nt * 16 + lm;
      const float v = acc[nt][r] + bias[gn];
      const int which = gn / 192;
      const int head  = (gn % 192) >> 5;
      const int hd    = gn & 31;
      const long win = gm / LW, t = gm % LW;
      const long di = ((win * HEADS + head) * LW + t) * HD + hd;
      if      (which == 0) q16[di] = (_Float16)(v * SCALE);
      else if (which == 1) k16[di] = (_Float16)v;
      else                 v16[di] = (_Float16)v;
    }
}

// ---------------- attention: 1 workgroup per (window, head), 9 waves ----------
__global__ __launch_bounds__(288) void attn_kernel(
    const _Float16* __restrict__ q16, const _Float16* __restrict__ k16,
    const _Float16* __restrict__ v16, const float* __restrict__ bias_tab,
    const int* __restrict__ pos_index, const float* __restrict__ attn_mask,
    _Float16* __restrict__ o16, int roll)
{
  __shared__ _Float16 P[144 * 160];   // softmax probs, K padded 144->160
  __shared__ _Float16 vT[32 * 160];   // v transposed [n][k], zero padded
  const int wg   = blockIdx.x;
  const int win  = wg / HEADS, head = wg % HEADS;
  const int wtyp = win & 63;
  const int tid  = threadIdx.x, lane = tid & 31, wave = tid >> 5;
  const int half = lane >> 4, lm = lane & 15;
  const long base = ((long)win * HEADS + head) * LW * HD;
  const _Float16* qb = q16 + base;
  const _Float16* kb = k16 + base;
  const _Float16* vb = v16 + base;

  for (int i = tid; i < 32 * 160; i += 288) {
    const int n = i / 160, kk = i % 160;
    vT[n * 160 + kk] = (kk < LW) ? vb[(long)kk * HD + n] : (_Float16)0.f;
  }
  __syncthreads();

  const int mbase = wave * 16;
  // S = q @ k^T  (M=16 rows per wave, K=32, N=144 => 9 wmma)
  v8h a0 = *(const v8h*)(qb + (long)(mbase + lm) * HD + half * 8);
  v8h a1 = *(const v8h*)(qb + (long)(mbase + lm) * HD + 16 + half * 8);
  v16h qa = cat16(a0, a1);
  v8f z8 = {0.f,0.f,0.f,0.f,0.f,0.f,0.f,0.f};
  v8f sacc[9];
  #pragma unroll
  for (int nj = 0; nj < 9; ++nj) {
    const int n = nj * 16 + lm;
    v8h b0 = *(const v8h*)(kb + (long)n * HD + half * 8);
    v8h b1 = *(const v8h*)(kb + (long)n * HD + 16 + half * 8);
    v16h kbv = cat16(b0, b1);
    sacc[nj] = __builtin_amdgcn_wmma_f32_16x16x32_f16(
        false, qa, false, kbv, (short)0, z8, false, false);
  }
  // + earth-specific bias (+ shift mask)
  #pragma unroll
  for (int nj = 0; nj < 9; ++nj)
    #pragma unroll
    for (int r = 0; r < 8; ++r) {
      const int mm = mbase + r + 8 * half;
      const int nn = nj * 16 + lm;
      const int pi = pos_index[mm * LW + nn];
      float sv = sacc[nj][r] + bias_tab[(long)pi * (64 * HEADS) + wtyp * HEADS + head];
      if (roll) sv += attn_mask[((long)win * LW + mm) * LW + nn];
      sacc[nj][r] = sv;
    }
  // row softmax (rows live in a half-wave; reduce across 16 lanes)
  float rmax[8], rinv[8];
  #pragma unroll
  for (int r = 0; r < 8; ++r) {
    float t = sacc[0][r];
    #pragma unroll
    for (int nj = 1; nj < 9; ++nj) t = fmaxf(t, sacc[nj][r]);
    #pragma unroll
    for (int off = 8; off >= 1; off >>= 1) t = fmaxf(t, __shfl_xor(t, off, 32));
    rmax[r] = t;
  }
  #pragma unroll
  for (int nj = 0; nj < 9; ++nj)
    #pragma unroll
    for (int r = 0; r < 8; ++r)
      sacc[nj][r] = __expf(sacc[nj][r] - rmax[r]);
  #pragma unroll
  for (int r = 0; r < 8; ++r) {
    float t = 0.f;
    #pragma unroll
    for (int nj = 0; nj < 9; ++nj) t += sacc[nj][r];
    #pragma unroll
    for (int off = 8; off >= 1; off >>= 1) t += __shfl_xor(t, off, 32);
    rinv[r] = 1.f / t;
  }
  #pragma unroll
  for (int nj = 0; nj < 9; ++nj)
    #pragma unroll
    for (int r = 0; r < 8; ++r) {
      const int mm = mbase + r + 8 * half;
      const int nn = nj * 16 + lm;
      P[mm * 160 + nn] = (_Float16)(sacc[nj][r] * rinv[r]);
    }
  for (int i = lane; i < 256; i += 32)     // zero the K padding cols
    P[(mbase + (i >> 4)) * 160 + 144 + (i & 15)] = (_Float16)0.f;
  __syncthreads();

  // O = P @ v   (K=160 padded => 5 ksteps, N=32 => 2 ntiles)
  v8f oacc[2] = {z8, z8};
  #pragma unroll
  for (int ks = 0; ks < 5; ++ks) {
    v8h p0 = *(const v8h*)&P[(mbase + lm) * 160 + ks * 32 + half * 8];
    v8h p1 = *(const v8h*)&P[(mbase + lm) * 160 + ks * 32 + 16 + half * 8];
    v16h pa = cat16(p0, p1);
    #pragma unroll
    for (int nt = 0; nt < 2; ++nt) {
      const int n = nt * 16 + lm;
      v8h vv0 = *(const v8h*)&vT[n * 160 + ks * 32 + half * 8];
      v8h vv1 = *(const v8h*)&vT[n * 160 + ks * 32 + 16 + half * 8];
      v16h vbv = cat16(vv0, vv1);
      oacc[nt] = __builtin_amdgcn_wmma_f32_16x16x32_f16(
          false, pa, false, vbv, (short)0, oacc[nt], false, false);
    }
  }
  #pragma unroll
  for (int nt = 0; nt < 2; ++nt)
    #pragma unroll
    for (int r = 0; r < 8; ++r) {
      const int mm = mbase + r + 8 * half;
      const int c  = head * HD + nt * 16 + lm;
      o16[((long)win * LW + mm) * Cc + c] = (_Float16)oacc[nt][r];
    }
}

// ---------------- proj GEMM + reverse-window + residual -----------------------
__global__ __launch_bounds__(256) void gemm_proj_kernel(
    const _Float16* __restrict__ A, const float* __restrict__ W,
    const float* __restrict__ bias, const float* __restrict__ x_old,
    float* __restrict__ x_new, int roll)
{
  __shared__ _Float16 As[128 * 32];
  __shared__ _Float16 Bs[64 * 32];
  const int m0 = blockIdx.x * 128, n0 = blockIdx.y * 64;
  v8f z8 = {0.f,0.f,0.f,0.f,0.f,0.f,0.f,0.f};
  v8f acc[4] = {z8, z8, z8, z8};
  gemm_mainloop<192, 192>(A, W, m0, n0, As, Bs, acc);
  const int tid = threadIdx.x, lane = tid & 31, wave = tid >> 5;
  const int half = lane >> 4, lm = lane & 15;
  #pragma unroll
  for (int nt = 0; nt < 4; ++nt)
    #pragma unroll
    for (int r = 0; r < 8; ++r) {
      const long gm = m0 + wave * 16 + r + 8 * half;
      const int  gn = n0 + nt * 16 + lm;
      const int win = (int)(gm / LW), t = (int)(gm % LW);
      const long tok = win_token(win, t, roll);
      x_new[tok * Cc + gn] = x_old[tok * Cc + gn] + acc[nt][r] + bias[gn];
    }
}

// ---------------- MLP GEMM 1: (NT,192)@(192,768)+b, exact GELU ---------------
__global__ __launch_bounds__(256) void gemm_mlp1_kernel(
    const _Float16* __restrict__ A, const float* __restrict__ W,
    const float* __restrict__ bias, _Float16* __restrict__ mid16)
{
  __shared__ _Float16 As[128 * 32];
  __shared__ _Float16 Bs[64 * 32];
  const int m0 = blockIdx.x * 128, n0 = blockIdx.y * 64;
  v8f z8 = {0.f,0.f,0.f,0.f,0.f,0.f,0.f,0.f};
  v8f acc[4] = {z8, z8, z8, z8};
  gemm_mainloop<192, 768>(A, W, m0, n0, As, Bs, acc);
  const int tid = threadIdx.x, lane = tid & 31, wave = tid >> 5;
  const int half = lane >> 4, lm = lane & 15;
  #pragma unroll
  for (int nt = 0; nt < 4; ++nt)
    #pragma unroll
    for (int r = 0; r < 8; ++r) {
      const long gm = m0 + wave * 16 + r + 8 * half;
      const int  gn = n0 + nt * 16 + lm;
      const float u = acc[nt][r] + bias[gn];
      const float gl = 0.5f * u * (1.f + erff(u * 0.7071067811865475f));
      mid16[gm * 768 + gn] = (_Float16)gl;
    }
}

// ---------------- MLP GEMM 2: (NT,768)@(768,192)+b, residual accumulate ------
__global__ __launch_bounds__(256) void gemm_mlp2_kernel(
    const _Float16* __restrict__ A, const float* __restrict__ W,
    const float* __restrict__ bias, float* __restrict__ xio)
{
  __shared__ _Float16 As[128 * 32];
  __shared__ _Float16 Bs[64 * 32];
  const int m0 = blockIdx.x * 128, n0 = blockIdx.y * 64;
  v8f z8 = {0.f,0.f,0.f,0.f,0.f,0.f,0.f,0.f};
  v8f acc[4] = {z8, z8, z8, z8};
  gemm_mainloop<768, 192>(A, W, m0, n0, As, Bs, acc);
  const int tid = threadIdx.x, lane = tid & 31, wave = tid >> 5;
  const int half = lane >> 4, lm = lane & 15;
  #pragma unroll
  for (int nt = 0; nt < 4; ++nt)
    #pragma unroll
    for (int r = 0; r < 8; ++r) {
      const long gm = m0 + wave * 16 + r + 8 * half;
      const int  gn = n0 + nt * 16 + lm;
      float* p = xio + gm * Cc + gn;
      *p = *p + acc[nt][r] + bias[gn];
    }
}

// ---------------- host driver ----------------
extern "C" void kernel_launch(void* const* d_in, const int* in_sizes, int n_in,
                              void* d_out, int out_size, void* d_ws, size_t ws_size,
                              hipStream_t stream) {
  const float* x_in    = (const float*)d_in[0];
  const float* qkv_w   = (const float*)d_in[1];
  const float* qkv_b   = (const float*)d_in[2];
  const float* proj_w  = (const float*)d_in[3];
  const float* proj_b  = (const float*)d_in[4];
  const float* bias_tb = (const float*)d_in[5];
  const float* n1_g    = (const float*)d_in[6];
  const float* n1_b    = (const float*)d_in[7];
  const float* n2_g    = (const float*)d_in[8];
  const float* n2_b    = (const float*)d_in[9];
  const float* mlp_w1  = (const float*)d_in[10];
  const float* mlp_b1  = (const float*)d_in[11];
  const float* mlp_w2  = (const float*)d_in[12];
  const float* mlp_b2  = (const float*)d_in[13];
  const int*   pos_idx = (const int*)  d_in[14];
  const float* a_mask  = (const float*)d_in[15];
  float* out = (float*)d_out;

  _Float16* h16  = (_Float16*)d_ws;
  _Float16* q16  = h16  + NT * Cc;
  _Float16* k16  = q16  + NT * Cc;
  _Float16* v16  = k16  + NT * Cc;
  _Float16* o16  = v16  + NT * Cc;
  _Float16* mid16 = q16;          // reuse q/k/v/o region (NT*768 halves)

  const int LNBLKS = (int)(NT / 8);         // 18432
  const int MTILES = (int)(NT / 128);       // 1152

  for (int blk = 0; blk < 2; ++blk) {
    const int roll = blk & 1;
    const float* xsrc = (blk == 0) ? x_in : out;

    ln1_win_kernel<<<LNBLKS, 256, 0, stream>>>(
        xsrc, n1_g + blk * Cc, n1_b + blk * Cc, h16, roll);

    gemm_qkv_kernel<<<dim3(MTILES, 9), 256, 0, stream>>>(
        h16, qkv_w + (long)blk * Cc * 576, qkv_b + blk * 576, q16, k16, v16);

    attn_kernel<<<NWIN * HEADS, 288, 0, stream>>>(
        q16, k16, v16, bias_tb + (long)blk * 3312 * 64 * HEADS,
        pos_idx, a_mask, o16, roll);

    gemm_proj_kernel<<<dim3(MTILES, 3), 256, 0, stream>>>(
        o16, proj_w + (long)blk * Cc * Cc, proj_b + blk * Cc, xsrc, out, roll);

    ln2_kernel<<<LNBLKS, 256, 0, stream>>>(
        out, n2_g + blk * Cc, n2_b + blk * Cc, h16);

    gemm_mlp1_kernel<<<dim3(MTILES, 12), 256, 0, stream>>>(
        h16, mlp_w1 + (long)blk * Cc * 768, mlp_b1 + blk * 768, mid16);

    gemm_mlp2_kernel<<<dim3(MTILES, 3), 256, 0, stream>>>(
        mid16, mlp_w2 + (long)blk * 768 * Cc, mlp_b2 + blk * Cc, out);
  }
}
